// Attention_44555990728894
// MI455X (gfx1250) — matrix operand order repaired and checked by reading, not compiled, not run on hardware
//
#include <hip/hip_runtime.h>

// ---- problem constants (match reference setup) ----
#define HQ 32
#define KVH 8
#define HD 128
#define REP (HQ / KVH)
#define CHUNK_START 2048
#define LOG2E 1.44269504088896340736f
#define QK_SCALE 0.08838834764831845f   // 1/sqrt(128)

typedef __attribute__((ext_vector_type(16))) _Float16 v16h;
typedef __attribute__((ext_vector_type(8)))  _Float16 v8h;
typedef __attribute__((ext_vector_type(8)))  float    v8f;
typedef __attribute__((ext_vector_type(4)))  float    v4f;

// ----------------------------------------------------------------------------
// Kernel 1: fill f16 workspace from the caches.
//   Kf  : [KVH][T][HD]  row-major f16  (for QK A-fragments)
//   Vt  : [KVH][HD][T]  transposed f16 (for PV A-fragments, V^T)
// ----------------------------------------------------------------------------
__global__ void prep_fill_kernel(const float* __restrict__ k_cache,
                                 const float* __restrict__ v_cache,
                                 _Float16* __restrict__ Kf,
                                 _Float16* __restrict__ Vt,
                                 int T) {
    int gid = blockIdx.x * blockDim.x + threadIdx.x;
    int total = KVH * T * HD;
    if (gid >= total) return;

    // K: destination-layout decomposition [kvh][t][d]
    {
        int kvh = gid / (T * HD);
        int rem = gid - kvh * T * HD;
        int t   = rem / HD;
        int d   = rem - t * HD;
        Kf[gid] = (_Float16)k_cache[((size_t)t * KVH + kvh) * HD + d];
    }
    // V^T: destination-layout decomposition [kvh][d][t]
    {
        int kvh = gid / (HD * T);
        int rem = gid - kvh * HD * T;
        int d   = rem / T;
        int t   = rem - d * T;
        Vt[gid] = (_Float16)v_cache[((size_t)t * KVH + kvh) * HD + d];
    }
}

// ----------------------------------------------------------------------------
// Kernel 2: scatter the new chunk's k/v rows via slot_mapping.
// ----------------------------------------------------------------------------
__global__ void prep_scatter_kernel(const float* __restrict__ k,
                                    const float* __restrict__ v,
                                    const int* __restrict__ slot_mapping,
                                    _Float16* __restrict__ Kf,
                                    _Float16* __restrict__ Vt,
                                    int Nq, int T) {
    int gid = blockIdx.x * blockDim.x + threadIdx.x;
    int total = Nq * KVH * HD;
    if (gid >= total) return;
    int i   = gid / (KVH * HD);
    int rem = gid - i * KVH * HD;
    int kvh = rem / HD;
    int d   = rem - kvh * HD;
    int slot = slot_mapping[i];
    if (slot < 0 || slot >= T) return;
    Kf[((size_t)kvh * T + slot) * HD + d]  = (_Float16)k[gid];
    Vt[((size_t)kvh * HD + d) * T + slot]  = (_Float16)v[gid];
}

// ----------------------------------------------------------------------------
// Kernel 3: flash attention, one wave per (head, 16-query tile).
//   S^T = K_tile * Q^T   via v_wmma_f32_16x16x32_f16  (A=K, B=Q^T, D=S^T)
//   O^T += V^T * P^T     via v_wmma_f32_16x16x32_f16  (A=V^T, B=P^T, D=O^T)
// S^T orientation puts each query's softmax row in a single lane pair:
// lane l holds S^T[key = kb + r + 8*(l>>4)][query = l&15] in acc[r].
//
// Main loop: 64 keys/iter, no causal mask (all keys provably valid).
// Tail loop: 32 keys/iter with per-element causal mask (<= 3 iters).
// ----------------------------------------------------------------------------
__global__ __launch_bounds__(128)
void attn_wmma_kernel(const float* __restrict__ q,
                      const _Float16* __restrict__ Kf,
                      const _Float16* __restrict__ Vt,
                      float* __restrict__ out,
                      int Nq, int T) {
    const int h    = blockIdx.x;
    const int kvh  = h / REP;
    const int wave = threadIdx.x >> 5;
    const int lane = threadIdx.x & 31;
    const int lmod = lane & 15;
    const int lhi  = lane >> 4;          // 0: lo half, 1: hi half
    const bool hiHalf = (lhi != 0);
    const int q0   = blockIdx.y * 64 + wave * 16;

    // ---- load Q^T B-fragments: 4 dim-chunks of 32 ----
    // B layout: element e of lane l holds B[Krow = e + 16*(l>>4)][N = l&15]
    v16h bq[4];
    {
        const float* qrow = q + ((size_t)(q0 + lmod) * HQ + h) * HD;
        #pragma unroll
        for (int c = 0; c < 4; ++c) {
            const float* p = qrow + 32 * c + 16 * lhi;
            v4f f0 = ((const v4f*)p)[0];
            v4f f1 = ((const v4f*)p)[1];
            v4f f2 = ((const v4f*)p)[2];
            v4f f3 = ((const v4f*)p)[3];
            v16h b;
            #pragma unroll
            for (int j = 0; j < 4; ++j) {
                b[j]      = (_Float16)f0[j];
                b[j + 4]  = (_Float16)f1[j];
                b[j + 8]  = (_Float16)f2[j];
                b[j + 12] = (_Float16)f3[j];
            }
            bq[c] = b;
        }
    }

    // ---- accumulators: O^T, 8 chunks of 16 dims; per-query m / l stats ----
    v8f ot[8];
    #pragma unroll
    for (int dc = 0; dc < 8; ++dc)
        ot[dc] = (v8f){0.f, 0.f, 0.f, 0.f, 0.f, 0.f, 0.f, 0.f};
    float m = -1e30f;
    float l = 0.f;

    const _Float16* kbase0 = Kf + ((size_t)kvh * T + lmod) * HD + 16 * lhi;
    const _Float16* vbase0 = Vt + ((size_t)kvh * HD + lmod) * T + 16 * lhi;

    const int qlim = CHUNK_START + q0 + lmod;  // this lane's query key-limit
    // keys [0, bulk_steps*64) are valid for EVERY query of this wave
    const int bulk_steps = (CHUNK_START + q0 + 1) / 64;
    const int tile_lim   = CHUNK_START + q0 + 15;

    // ================= bulk: 64 keys per iteration, no mask =================
    for (int it = 0; it < bulk_steps; ++it) {
        const int t0 = it * 64;

        // ---- QK: four 16-key subtiles, each over 4 dim-chunks ----
        float s[4][8];
        #pragma unroll
        for (int sub = 0; sub < 4; ++sub) {
            v8f acc = (v8f){0.f, 0.f, 0.f, 0.f, 0.f, 0.f, 0.f, 0.f};
            const _Float16* krow = kbase0 + (size_t)(t0 + 16 * sub) * HD;
            #pragma unroll
            for (int c = 0; c < 4; ++c) {
                v8h lo = *(const v8h*)(krow + 32 * c);
                v8h hi = *(const v8h*)(krow + 32 * c + 8);
                v16h a = __builtin_shufflevector(lo, hi,
                    0, 1, 2, 3, 4, 5, 6, 7, 8, 9, 10, 11, 12, 13, 14, 15);
                acc = __builtin_amdgcn_wmma_f32_16x16x32_f16(
                    false, a, false, bq[c], (short)0, acc, false, false);
            }
            #pragma unroll
            for (int r = 0; r < 8; ++r)
                s[sub][r] = acc[r] * QK_SCALE;
        }

        // ---- online softmax over the 64 keys ----
        float tmax = -1e30f;
        #pragma unroll
        for (int sub = 0; sub < 4; ++sub)
            #pragma unroll
            for (int r = 0; r < 8; ++r)
                tmax = fmaxf(tmax, s[sub][r]);
        tmax = fmaxf(tmax, __shfl_xor(tmax, 16, 32));
        const float newm  = fmaxf(m, tmax);
        const float alpha = __builtin_amdgcn_exp2f((m - newm) * LOG2E);

        float rsum = 0.f;
        #pragma unroll
        for (int sub = 0; sub < 4; ++sub)
            #pragma unroll
            for (int r = 0; r < 8; ++r) {
                s[sub][r] = __builtin_amdgcn_exp2f((s[sub][r] - newm) * LOG2E);
                rsum += s[sub][r];
            }
        rsum += __shfl_xor(rsum, 16, 32);
        l = l * alpha + rsum;
        m = newm;

        #pragma unroll
        for (int dc = 0; dc < 8; ++dc)
            #pragma unroll
            for (int r = 0; r < 8; ++r)
                ot[dc][r] *= alpha;

        // ---- build two P^T B-fragments (32 keys each) ----
        v16h bp[2];
        #pragma unroll
        for (int g = 0; g < 2; ++g) {
            #pragma unroll
            for (int r = 0; r < 8; ++r) {
                float x0p = __shfl_xor(s[2 * g][r], 16, 32);
                float x1p = __shfl_xor(s[2 * g + 1][r], 16, 32);
                bp[g][r]     = hiHalf ? (_Float16)x1p : (_Float16)s[2 * g][r];
                bp[g][r + 8] = hiHalf ? (_Float16)s[2 * g + 1][r] : (_Float16)x0p;
            }
        }

        // ---- PV: O^T += V^T * P^T, two 32-key groups ----
        #pragma unroll
        for (int g = 0; g < 2; ++g) {
            const _Float16* vb = vbase0 + t0 + 32 * g;
            #pragma unroll
            for (int dc = 0; dc < 8; ++dc) {
                const _Float16* vrow = vb + (size_t)(16 * dc) * T;
                v8h lo = *(const v8h*)(vrow);
                v8h hi = *(const v8h*)(vrow + 8);
                v16h a = __builtin_shufflevector(lo, hi,
                    0, 1, 2, 3, 4, 5, 6, 7, 8, 9, 10, 11, 12, 13, 14, 15);
                ot[dc] = __builtin_amdgcn_wmma_f32_16x16x32_f16(
                    false, a, false, bp[g], (short)0, ot[dc], false, false);
            }
        }
    }

    // ================= tail: 32 keys per iteration, masked =================
    for (int t0 = bulk_steps * 64; t0 <= tile_lim; t0 += 32) {
        float s0[8], s1[8];
        #pragma unroll
        for (int sub = 0; sub < 2; ++sub) {
            const int kb = t0 + 16 * sub;
            v8f acc = (v8f){0.f, 0.f, 0.f, 0.f, 0.f, 0.f, 0.f, 0.f};
            const _Float16* krow = kbase0 + (size_t)kb * HD;
            #pragma unroll
            for (int c = 0; c < 4; ++c) {
                v8h lo = *(const v8h*)(krow + 32 * c);
                v8h hi = *(const v8h*)(krow + 32 * c + 8);
                v16h a = __builtin_shufflevector(lo, hi,
                    0, 1, 2, 3, 4, 5, 6, 7, 8, 9, 10, 11, 12, 13, 14, 15);
                acc = __builtin_amdgcn_wmma_f32_16x16x32_f16(
                    false, a, false, bq[c], (short)0, acc, false, false);
            }
            float* sd = sub ? s1 : s0;
            #pragma unroll
            for (int r = 0; r < 8; ++r) {
                int key = kb + r + 8 * lhi;
                float sv = acc[r] * QK_SCALE;
                sd[r] = (key <= qlim) ? sv : -1e30f;
            }
        }

        float tmax = -1e30f;
        #pragma unroll
        for (int r = 0; r < 8; ++r)
            tmax = fmaxf(tmax, fmaxf(s0[r], s1[r]));
        tmax = fmaxf(tmax, __shfl_xor(tmax, 16, 32));
        const float newm  = fmaxf(m, tmax);
        const float alpha = __builtin_amdgcn_exp2f((m - newm) * LOG2E);

        float rsum = 0.f;
        #pragma unroll
        for (int r = 0; r < 8; ++r) {
            s0[r] = __builtin_amdgcn_exp2f((s0[r] - newm) * LOG2E);
            s1[r] = __builtin_amdgcn_exp2f((s1[r] - newm) * LOG2E);
            rsum += s0[r] + s1[r];
        }
        rsum += __shfl_xor(rsum, 16, 32);
        l = l * alpha + rsum;
        m = newm;

        #pragma unroll
        for (int dc = 0; dc < 8; ++dc)
            #pragma unroll
            for (int r = 0; r < 8; ++r)
                ot[dc][r] *= alpha;

        v16h bp;
        #pragma unroll
        for (int r = 0; r < 8; ++r) {
            float x0p = __shfl_xor(s0[r], 16, 32);
            float x1p = __shfl_xor(s1[r], 16, 32);
            bp[r]     = hiHalf ? (_Float16)x1p : (_Float16)s0[r];
            bp[r + 8] = hiHalf ? (_Float16)s1[r] : (_Float16)x0p;
        }

        const _Float16* vb = vbase0 + t0;
        #pragma unroll
        for (int dc = 0; dc < 8; ++dc) {
            const _Float16* vrow = vb + (size_t)(16 * dc) * T;
            v8h lo = *(const v8h*)(vrow);
            v8h hi = *(const v8h*)(vrow + 8);
            v16h a = __builtin_shufflevector(lo, hi,
                0, 1, 2, 3, 4, 5, 6, 7, 8, 9, 10, 11, 12, 13, 14, 15);
            ot[dc] = __builtin_amdgcn_wmma_f32_16x16x32_f16(
                false, a, false, bp, (short)0, ot[dc], false, false);
        }
    }

    // ---- epilogue: normalize and store; dims are contiguous per lane ----
    const float inv = 1.0f / l;
    float* orow = out + ((size_t)(q0 + lmod) * HQ + h) * HD + 8 * lhi;
    #pragma unroll
    for (int dc = 0; dc < 8; ++dc) {
        v4f a, b;
        #pragma unroll
        for (int j = 0; j < 4; ++j) {
            a[j] = ot[dc][j] * inv;
            b[j] = ot[dc][j + 4] * inv;
        }
        *(v4f*)(orow + 16 * dc)     = a;
        *(v4f*)(orow + 16 * dc + 4) = b;
    }
}

// ----------------------------------------------------------------------------
// host launcher
// ----------------------------------------------------------------------------
extern "C" void kernel_launch(void* const* d_in, const int* in_sizes, int n_in,
                              void* d_out, int out_size, void* d_ws, size_t ws_size,
                              hipStream_t stream) {
    const float* q        = (const float*)d_in[0];
    const float* k        = (const float*)d_in[1];
    const float* v        = (const float*)d_in[2];
    const float* k_cache  = (const float*)d_in[3];
    const float* v_cache  = (const float*)d_in[4];
    const int*   slot_map = (const int*)d_in[5];
    float*       out      = (float*)d_out;

    const int Nq = in_sizes[0] / (HQ * HD);   // 2048
    const int T  = CHUNK_START + Nq;          // 4096

    // workspace: Kf [KVH][T][HD] f16, then Vt [KVH][HD][T] f16
    _Float16* Kf = (_Float16*)d_ws;
    _Float16* Vt = Kf + (size_t)KVH * T * HD;

    {
        int total = KVH * T * HD;
        prep_fill_kernel<<<(total + 255) / 256, 256, 0, stream>>>(
            k_cache, v_cache, Kf, Vt, T);
    }
    {
        int total = Nq * KVH * HD;
        prep_scatter_kernel<<<(total + 255) / 256, 256, 0, stream>>>(
            k, v, slot_map, Kf, Vt, Nq, T);
    }
    {
        dim3 grid(HQ, Nq / 64);
        attn_wmma_kernel<<<grid, 128, 0, stream>>>(q, Kf, Vt, out, Nq, T);
    }
}
